// MultiHeadAttention_47708496724532
// MI455X (gfx1250) — matrix operand order repaired and checked by reading, not compiled, not run on hardware
//
#include <hip/hip_runtime.h>
#include <hip/hip_bf16.h>
#include <math.h>

// ---------------------------------------------------------------------------
// Types for CDNA5 WMMA (wave32): v_wmma_f32_16x16x32_bf16
// ---------------------------------------------------------------------------
typedef __bf16 bf16_t;
typedef __attribute__((ext_vector_type(4)))  bf16_t v4bf;
typedef __attribute__((ext_vector_type(8)))  bf16_t v8bf;
typedef __attribute__((ext_vector_type(16))) bf16_t v16bf;
typedef __attribute__((ext_vector_type(8)))  float  v8f;

#define CDIM   256
#define NGRAPH 512
#define ROWS   128          // nodes per workgroup
#define RT     (ROWS / 16)  // 8 row tiles of 16

// ---------------------------------------------------------------------------
// Compute one 16-wide column tile of D = A(ROWSx256) * W(256x16) for all 8
// row tiles at once: each weight B-fragment (one 32B contiguous load from the
// transposed bf16 weight) feeds 8 independent WMMA accumulator chains.
// A-fragment (16-bit A 16x32): lane m=lane&15, half kh=lane>>4,
//   elems 0..15 = K kb+kh*16 .. +16
// B-fragment (32x16): lane col n=lane&15, rows K = kb+kh*16 .. +16
// ---------------------------------------------------------------------------
__device__ __forceinline__ void wmma_block_coltile(
    const bf16_t* __restrict__ sA,   // LDS [ROWS][256] bf16
    const bf16_t* __restrict__ WT,   // global bf16, transposed [n][k]
    int tile, int lane, v8f acc[RT])
{
    const int m   = lane & 15;
    const int kh  = lane >> 4;
    const int col = tile * 16 + (lane & 15);
#pragma unroll 2
    for (int ks = 0; ks < 8; ++ks) {
        const int kb = ks * 32;
        const v16bf b = *(const v16bf*)(WT + (size_t)col * CDIM + kb + kh * 16);
#pragma unroll
        for (int r = 0; r < RT; ++r) {
            const bf16_t* arow = sA + (r * 16 + m) * CDIM + kb + kh * 16;
            v16bf a = *(const v16bf*)(arow);
            acc[r] = __builtin_amdgcn_wmma_f32_16x16x32_bf16(
                        false, a, false, b, (short)0, acc[r], false, false);
        }
    }
}

// ordered-int float atomic max (segMax pre-initialized to -inf bit pattern)
__device__ __forceinline__ void atomicMaxF(float* addr, float v)
{
    if (v >= 0.f) atomicMax((int*)addr, __float_as_int(v));
    else          atomicMin((unsigned int*)addr, __float_as_uint(v));
}

// ---------------------------------------------------------------------------
// K0: weights -> bf16 transposed [n][k]; init segment max/sum
// ---------------------------------------------------------------------------
__global__ void __launch_bounds__(256)
k_init(const float* __restrict__ Wq, const float* __restrict__ Wk,
       const float* __restrict__ Wv, const float* __restrict__ Wo,
       bf16_t* __restrict__ WT, float* __restrict__ segMax, float* __restrict__ segSum)
{
    const int t = blockIdx.x * 256 + threadIdx.x;
    if (t < 4 * CDIM * CDIM) {
        const int mat = t >> 16;
        const int idx = t & 65535;
        const int k = idx >> 8, n = idx & 255;
        const float* W = (mat == 0) ? Wq : (mat == 1) ? Wk : (mat == 2) ? Wv : Wo;
        WT[(size_t)mat * CDIM * CDIM + n * CDIM + k] = (bf16_t)W[k * CDIM + n];
    }
    if (t < NGRAPH * 16) {
        ((int*)segMax)[t] = 0xFF800000;   // -inf
        segSum[t] = 0.f;
    }
}

// ---------------------------------------------------------------------------
// K1: per 128-node tile: Q,K,V = x@W+b (bf16 WMMA), 4x4 per-node scores,
//     V -> ws (bf16), grouped atomic segment-max (batch is sorted)
// ---------------------------------------------------------------------------
__global__ void __launch_bounds__(256)
k_qkv_scores(const float* __restrict__ x, const int* __restrict__ batch,
             const bf16_t* __restrict__ WqT, const bf16_t* __restrict__ WkT,
             const bf16_t* __restrict__ WvT,
             const float* __restrict__ bq, const float* __restrict__ bk,
             const float* __restrict__ bv,
             bf16_t* __restrict__ Vws, float* __restrict__ scores,
             float* __restrict__ segMax, int n)
{
    __shared__ bf16_t sX[ROWS * CDIM];   // 64 KB
    __shared__ bf16_t sQ[ROWS * CDIM];   // 64 KB
    __shared__ bf16_t sK[ROWS * CDIM];   // 64 KB
    __shared__ float  sS[ROWS * 16];     //  8 KB
    __shared__ int    sB[ROWS];

    const int tid  = threadIdx.x;
    const int base = blockIdx.x * ROWS;

    for (int r = tid; r < ROWS; r += 256)
        sB[r] = (base + r < n) ? batch[base + r] : -1;

    // x tile -> bf16 LDS, float4 per lane
    {
        const int c4 = (tid & 63) * 4;
        const int r0 = tid >> 6;
#pragma unroll
        for (int pass = 0; pass < ROWS / 4; ++pass) {
            const int r   = r0 + pass * 4;
            const int row = base + r;
            float4 xv = make_float4(0.f, 0.f, 0.f, 0.f);
            if (row < n) xv = *(const float4*)(x + (size_t)row * CDIM + c4);
            v4bf p;
            p[0] = (bf16_t)xv.x; p[1] = (bf16_t)xv.y;
            p[2] = (bf16_t)xv.z; p[3] = (bf16_t)xv.w;
            *(v4bf*)(sX + r * CDIM + c4) = p;
        }
    }
    __syncthreads();

    const int lane = tid & 31;
    const int wv8  = tid >> 5;        // 8 waves, 2 column tiles each
    const int nlo  = lane & 15;
    const int mh   = lane >> 4;

#pragma unroll
    for (int tt = 0; tt < 2; ++tt) {
        const int tile = wv8 * 2 + tt;
        const int col  = tile * 16 + nlo;
        v8f acc[RT];

        // ---- Q ----
#pragma unroll
        for (int r = 0; r < RT; ++r) acc[r] = (v8f){};
        wmma_block_coltile(sX, WqT, tile, lane, acc);
        {
            const float bb = bq[col];
#pragma unroll
            for (int r = 0; r < RT; ++r)
#pragma unroll
                for (int p = 0; p < 8; ++p)
                    sQ[(r * 16 + p + 8 * mh) * CDIM + col] = (bf16_t)(acc[r][p] + bb);
        }
        // ---- K ----
#pragma unroll
        for (int r = 0; r < RT; ++r) acc[r] = (v8f){};
        wmma_block_coltile(sX, WkT, tile, lane, acc);
        {
            const float bb = bk[col];
#pragma unroll
            for (int r = 0; r < RT; ++r)
#pragma unroll
                for (int p = 0; p < 8; ++p)
                    sK[(r * 16 + p + 8 * mh) * CDIM + col] = (bf16_t)(acc[r][p] + bb);
        }
        // ---- V -> global ws (bf16) ----
#pragma unroll
        for (int r = 0; r < RT; ++r) acc[r] = (v8f){};
        wmma_block_coltile(sX, WvT, tile, lane, acc);
        {
            const float bb = bv[col];
#pragma unroll
            for (int r = 0; r < RT; ++r)
#pragma unroll
                for (int p = 0; p < 8; ++p) {
                    const int row = base + r * 16 + p + 8 * mh;
                    if (row < n)
                        Vws[(size_t)row * CDIM + col] = (bf16_t)(acc[r][p] + bb);
                }
        }
    }
    __syncthreads();

    // per-node head-vs-head scores: s[i][h][g] = q_h . k_g / 8
#pragma unroll
    for (int it = 0; it < (ROWS * 16) / 256; ++it) {
        const int idx = it * 256 + tid;
        const int i = idx >> 4, pr = idx & 15;
        const int h = pr >> 2,  g = pr & 3;
        float s = 0.f;
#pragma unroll
        for (int d = 0; d < 64; ++d)
            s += (float)sQ[i * CDIM + h * 64 + d] * (float)sK[i * CDIM + g * 64 + d];
        s *= 0.125f;   // 1/sqrt(64)
        sS[idx] = s;
        if (base + i < n) scores[(size_t)(base + i) * 16 + pr] = s;
    }
    __syncthreads();

    if (tid < 16) {   // one atomic per (graph, pair) per block
        int curg = -1; float m = -INFINITY;
        for (int i = 0; i < ROWS; ++i) {
            const int g = sB[i];
            if (g < 0) continue;
            const float s = sS[i * 16 + tid];
            if (g != curg) {
                if (curg >= 0) atomicMaxF(&segMax[curg * 16 + tid], m);
                curg = g; m = s;
            } else m = fmaxf(m, s);
        }
        if (curg >= 0) atomicMaxF(&segMax[curg * 16 + tid], m);
    }
}

// ---------------------------------------------------------------------------
// K2: e = exp(s - segMax), in place; grouped atomic segment-sum
// ---------------------------------------------------------------------------
__global__ void __launch_bounds__(256)
k_exp_segsum(const int* __restrict__ batch, float* __restrict__ scores,
             const float* __restrict__ segMax, float* __restrict__ segSum, int n)
{
    __shared__ float sS[ROWS * 16];
    __shared__ int   sB[ROWS];
    const int tid  = threadIdx.x;
    const int base = blockIdx.x * ROWS;

    for (int r = tid; r < ROWS; r += 256)
        sB[r] = (base + r < n) ? batch[base + r] : -1;

#pragma unroll
    for (int it = 0; it < (ROWS * 16) / 256; ++it) {
        const int idx = it * 256 + tid;
        const int i = idx >> 4, pr = idx & 15;
        const int row = base + i;
        float e = 0.f;
        if (row < n) {
            const int g = batch[row];
            e = __expf(scores[(size_t)row * 16 + pr] - segMax[g * 16 + pr]);
            scores[(size_t)row * 16 + pr] = e;
        }
        sS[idx] = e;
    }
    __syncthreads();

    if (tid < 16) {
        int curg = -1; float acc = 0.f;
        for (int j = 0; j < ROWS; ++j) {
            const int g = sB[j];
            if (g < 0) continue;
            const float v = sS[j * 16 + tid];
            if (g != curg) {
                if (curg >= 0) atomicAdd(&segSum[curg * 16 + tid], acc);
                curg = g; acc = v;
            } else acc += v;
        }
        if (curg >= 0) atomicAdd(&segSum[curg * 16 + tid], acc);
    }
}

// ---------------------------------------------------------------------------
// K3: att = e/sum; att.mean (output 2); y = att@v; out = y@Wo + bo (WMMA)
// ---------------------------------------------------------------------------
__global__ void __launch_bounds__(256)
k_out(const int* __restrict__ batch, const float* __restrict__ scores,
      const float* __restrict__ segSum, const bf16_t* __restrict__ Vws,
      const bf16_t* __restrict__ WoT, const float* __restrict__ bo,
      float* __restrict__ out, float* __restrict__ attMean, int n)
{
    __shared__ float  sAtt[ROWS * 16];   //  8 KB
    __shared__ bf16_t sV[ROWS * CDIM];   // 64 KB
    __shared__ bf16_t sY[ROWS * CDIM];   // 64 KB

    const int tid  = threadIdx.x;
    const int base = blockIdx.x * ROWS;

#pragma unroll
    for (int it = 0; it < (ROWS * 16) / 256; ++it) {
        const int idx = it * 256 + tid;
        const int i = idx >> 4, pr = idx & 15;
        const int row = base + i;
        float a = 0.f;
        if (row < n) {
            const int g = batch[row];
            a = scores[(size_t)row * 16 + pr] / (segSum[g * 16 + pr] + 1e-16f);
        }
        sAtt[idx] = a;
    }
    {   // V tile -> LDS (16B per lane per pass)
        const int c8 = (tid & 31) * 8;
        const int r0 = tid >> 5;
#pragma unroll
        for (int pass = 0; pass < ROWS / 8; ++pass) {
            const int r   = r0 + pass * 8;
            const int row = base + r;
            v8bf v = {};
            if (row < n) v = *(const v8bf*)(Vws + (size_t)row * CDIM + c8);
            *(v8bf*)(sV + r * CDIM + c8) = v;
        }
    }
    __syncthreads();

    // att.mean over heads axis -> [n, 4]
#pragma unroll
    for (int it = 0; it < (ROWS * 4) / 256; ++it) {
        const int idx = it * 256 + tid;
        const int i = idx >> 2, g = idx & 3;
        const int row = base + i;
        if (row < n) {
            float m = 0.f;
#pragma unroll
            for (int h = 0; h < 4; ++h) m += sAtt[i * 16 + h * 4 + g];
            attMean[(size_t)row * 4 + g] = m * 0.25f;
        }
    }

    {   // y[i][h*64+d] = sum_g att[i][h][g] * v[i][g*64+d]
        const int c = tid, h = c >> 6, d = c & 63;
        for (int i = 0; i < ROWS; ++i) {
            float y = 0.f;
#pragma unroll
            for (int g = 0; g < 4; ++g)
                y += sAtt[i * 16 + h * 4 + g] * (float)sV[i * CDIM + g * 64 + d];
            sY[i * CDIM + c] = (bf16_t)y;
        }
    }
    __syncthreads();

    const int lane = tid & 31, wv8 = tid >> 5;
    const int nlo = lane & 15, mh = lane >> 4;
#pragma unroll
    for (int tt = 0; tt < 2; ++tt) {
        const int tile = wv8 * 2 + tt;
        const int col  = tile * 16 + nlo;
        v8f acc[RT];
#pragma unroll
        for (int r = 0; r < RT; ++r) acc[r] = (v8f){};
        wmma_block_coltile(sY, WoT, tile, lane, acc);
        const float bb = bo[col];
#pragma unroll
        for (int r = 0; r < RT; ++r)
#pragma unroll
            for (int p = 0; p < 8; ++p) {
                const int row = base + r * 16 + p + 8 * mh;
                if (row < n) out[(size_t)row * CDIM + col] = acc[r][p] + bb;
            }
    }
}

// ---------------------------------------------------------------------------
extern "C" void kernel_launch(void* const* d_in, const int* in_sizes, int n_in,
                              void* d_out, int out_size, void* d_ws, size_t ws_size,
                              hipStream_t stream)
{
    const float* x     = (const float*)d_in[0];
    const int*   batch = (const int*)  d_in[1];
    const float* Wq    = (const float*)d_in[2];
    const float* bq    = (const float*)d_in[3];
    const float* Wk    = (const float*)d_in[4];
    const float* bk    = (const float*)d_in[5];
    const float* Wv    = (const float*)d_in[6];
    const float* bv    = (const float*)d_in[7];
    const float* Wo    = (const float*)d_in[8];
    const float* bo    = (const float*)d_in[9];
    const int n = in_sizes[0] / CDIM;

    // workspace layout
    char*   ws     = (char*)d_ws;
    bf16_t* WT     = (bf16_t*)ws;                                  // 512 KB
    float*  segMax = (float*)(ws + (size_t)4 * CDIM * CDIM * 2);   //  32 KB
    float*  segSum = segMax + NGRAPH * 16;                         //  32 KB
    float*  scores = segSum + NGRAPH * 16;                         // n*16 f32
    bf16_t* Vws    = (bf16_t*)(scores + (size_t)n * 16);           // n*256 bf16

    float* out     = (float*)d_out;
    float* attMean = out + (size_t)n * CDIM;

    const int blocks = (n + ROWS - 1) / ROWS;

    k_init<<<(4 * CDIM * CDIM + 255) / 256, 256, 0, stream>>>(Wq, Wk, Wv, Wo,
                                                              WT, segMax, segSum);
    k_qkv_scores<<<blocks, 256, 0, stream>>>(x, batch,
                                             WT, WT + CDIM * CDIM, WT + 2 * CDIM * CDIM,
                                             bq, bk, bv, Vws, scores, segMax, n);
    k_exp_segsum<<<blocks, 256, 0, stream>>>(batch, scores, segMax, segSum, n);
    k_out<<<blocks, 256, 0, stream>>>(batch, scores, segSum, Vws,
                                      WT + 3 * CDIM * CDIM, bo, out, attMean, n);
}
